// Model_39676907882196
// MI455X (gfx1250) — matrix operand order repaired and checked by reading, not compiled, not run on hardware
//
#include <hip/hip_runtime.h>
#include <math.h>

typedef __attribute__((ext_vector_type(16))) __bf16 v16bf;
typedef __attribute__((ext_vector_type(8)))  float  v8f;

constexpr int B_  = 8;
constexpr int S_  = 2048;   // Sq == Skv
constexpr int D_  = 128;    // D == Dv
constexpr int KVT = 32;     // kv tile (== WMMA K of GEMM2)
constexpr int QW  = 16;     // q rows per wave
constexpr int NW  = 4;      // waves per block
constexpr int QB  = QW * NW;

constexpr int KS = 144;     // k_lds row stride (bf16 elems): 288B -> rows shift 8 banks, conflict-free 32B reads
constexpr int VS = 40;      // vt_lds row stride
constexpr int PS = 40;      // p_lds row stride (16B reads conflict-free)

__device__ __forceinline__ __bf16 bf(float f) { return (__bf16)f; }  // native v_cvt, RNE

__device__ __forceinline__ v16bf ld_pair16(const __bf16* p0, const __bf16* p1) {
  union { uint4 u[2]; v16bf v; } r;
  r.u[0] = *(const uint4*)p0;
  r.u[1] = *(const uint4*)p1;
  return r.v;
}

// 16-lane XOR butterfly steps on VALU (v_permlane16_b32), no LDS traffic.
__device__ __forceinline__ float pl16(float x, int lo, int hi) {
  int i = __builtin_bit_cast(int, x);
  int r = __builtin_amdgcn_permlane16(i, i, lo, hi, false, false);
  return __builtin_bit_cast(float, r);
}
__device__ __forceinline__ float rowmax16(float x) {
  x = fmaxf(x, pl16(x, (int)0x67452301u, (int)0xEFCDAB89u));  // i^1
  x = fmaxf(x, pl16(x, (int)0x54761032u, (int)0xDCFE98BAu));  // i^2
  x = fmaxf(x, pl16(x, (int)0x32107654u, (int)0xBA98FEDCu));  // i^4
  x = fmaxf(x, pl16(x, (int)0xFEDCBA98u, (int)0x76543210u));  // i^8
  return x;
}
__device__ __forceinline__ float rowsum16(float x) {
  x += pl16(x, (int)0x67452301u, (int)0xEFCDAB89u);
  x += pl16(x, (int)0x54761032u, (int)0xDCFE98BAu);
  x += pl16(x, (int)0x32107654u, (int)0xBA98FEDCu);
  x += pl16(x, (int)0xFEDCBA98u, (int)0x76543210u);
  return x;
}

__global__ __launch_bounds__(128, 1)
void fa_fwd(const float* __restrict__ Q, const float* __restrict__ K,
            const float* __restrict__ V, const float* __restrict__ Msk,
            float* __restrict__ O) {
  __shared__ __bf16 k_lds[KVT * KS];      // K tile  [kv][d]  bf16
  __shared__ __bf16 vt_lds[D_ * VS];      // V tile  [dv][kv] bf16 (transposed)
  __shared__ __bf16 p_lds[NW * QW * PS];  // masked probs, per-wave

  const int tid  = threadIdx.x;
  const int lane = tid & 31;
  const int wv   = tid >> 5;
  const int hf   = lane >> 4;   // half-wave id
  const int ln   = lane & 15;
  const int b    = blockIdx.y;
  const int q0   = blockIdx.x * QB + wv * QW;

  // ---- Q tile -> 4 bf16 A-operands (contiguous K: 16*hf .. 16*hf+15) ----
  v16bf a_q[4];
  {
    const float* qrow = Q + ((size_t)b * S_ + (size_t)(q0 + ln)) * D_;
#pragma unroll
    for (int c = 0; c < 4; ++c) {
      const int d0 = 32 * c + 16 * hf;
      float4 f0 = *(const float4*)(qrow + d0);
      float4 f1 = *(const float4*)(qrow + d0 + 4);
      float4 f2 = *(const float4*)(qrow + d0 + 8);
      float4 f3 = *(const float4*)(qrow + d0 + 12);
      union { __bf16 h[16]; v16bf v; } pk;
      pk.h[0]=bf(f0.x);  pk.h[1]=bf(f0.y);  pk.h[2]=bf(f0.z);  pk.h[3]=bf(f0.w);
      pk.h[4]=bf(f1.x);  pk.h[5]=bf(f1.y);  pk.h[6]=bf(f1.z);  pk.h[7]=bf(f1.w);
      pk.h[8]=bf(f2.x);  pk.h[9]=bf(f2.y);  pk.h[10]=bf(f2.z); pk.h[11]=bf(f2.w);
      pk.h[12]=bf(f3.x); pk.h[13]=bf(f3.y); pk.h[14]=bf(f3.z); pk.h[15]=bf(f3.w);
      a_q[c] = pk.v;
    }
  }

  v8f zf;
#pragma unroll
  for (int i = 0; i < 8; ++i) zf[i] = 0.0f;

  v8f acc[8];                      // O accumulator: 8 dv-tiles of 16x16 f32
#pragma unroll
  for (int t = 0; t < 8; ++t) acc[t] = zf;
  float m_run[8], l_run[8];        // per-lane stats for its 8 C-layout rows
#pragma unroll
  for (int v = 0; v < 8; ++v) { m_run[v] = -INFINITY; l_run[v] = 0.0f; }

  for (int kv0 = 0; kv0 < S_; kv0 += KVT) {
    __syncthreads();               // previous tile fully consumed
    // ---- cooperative K/V tile load (fp32 -> bf16), V stored transposed ----
    {
      const int r  = tid >> 2;     // 0..31 kv row
      const int c4 = tid & 3;
      const float* krow = K + ((size_t)b * S_ + (size_t)(kv0 + r)) * D_;
      const float* vrow = V + ((size_t)b * S_ + (size_t)(kv0 + r)) * D_;
#pragma unroll
      for (int i = 0; i < 8; ++i) {
        const int col = (c4 + 4 * i) * 4;          // 0..124
        float4 kf = *(const float4*)(krow + col);
        union { __bf16 h[4]; uint2 u; } kp;
        kp.h[0]=bf(kf.x); kp.h[1]=bf(kf.y); kp.h[2]=bf(kf.z); kp.h[3]=bf(kf.w);
        *(uint2*)&k_lds[r * KS + col] = kp.u;
        float4 vf = *(const float4*)(vrow + col);
        vt_lds[(col + 0) * VS + r] = bf(vf.x);
        vt_lds[(col + 1) * VS + r] = bf(vf.y);
        vt_lds[(col + 2) * VS + r] = bf(vf.z);
        vt_lds[(col + 3) * VS + r] = bf(vf.w);
      }
    }
    // ---- dropout mask loads (coalesced; dominant HBM traffic) ----
    float mk0[8], mk1[8];
#pragma unroll
    for (int v = 0; v < 8; ++v) {
      const float* mr = Msk + ((size_t)b * S_ + (size_t)(q0 + 8 * hf + v)) * (size_t)S_ + kv0;
      mk0[v] = mr[ln];
      mk1[v] = mr[ln + 16];
    }
    // prefetch next tile's mask rows (global_prefetch_b8); 32 lanes cover 16 rows x 128B
    if (kv0 + KVT < S_) {
      const float* pf = Msk + ((size_t)b * S_ + (size_t)(q0 + ln)) * (size_t)S_
                      + (size_t)(kv0 + KVT + 16 * hf);
      __builtin_prefetch(pf, 0, 3);
    }
    __syncthreads();               // K/V tile visible

    // ---- GEMM1: S = Q @ K^T. Preload ALL B operands (16 ds_load_b128 in
    //      flight), then run the WMMA chains with partial dscnt waits. ----
    v16bf bk[8];
#pragma unroll
    for (int c = 0; c < 4; ++c) {
      const __bf16* kr0 = &k_lds[ln * KS + 32 * c + 16 * hf];
      const __bf16* kr1 = &k_lds[(16 + ln) * KS + 32 * c + 16 * hf];
      bk[2 * c]     = ld_pair16(kr0, kr0 + 8);   // B: contiguous 16 K per half-wave
      bk[2 * c + 1] = ld_pair16(kr1, kr1 + 8);
    }
    v8f s0 = zf, s1 = zf;
#pragma unroll
    for (int c = 0; c < 4; ++c) {
      s0 = __builtin_amdgcn_wmma_f32_16x16x32_bf16(false, a_q[c], false, bk[2*c],   (short)0, s0, false, false);
      s1 = __builtin_amdgcn_wmma_f32_16x16x32_bf16(false, a_q[c], false, bk[2*c+1], (short)0, s1, false, false);
    }

    // ---- issue GEMM2 B operands now; they complete under the softmax ----
    v16bf bv[8];
#pragma unroll
    for (int t = 0; t < 8; ++t) {
      const __bf16* vr = &vt_lds[(16 * t + ln) * VS + 16 * hf];
      bv[t] = ld_pair16(vr, vr + 8);
    }

    // ---- online softmax (row reductions via v_permlane16, pure VALU) ----
    float p0[8], p1[8], sc[8];
#pragma unroll
    for (int v = 0; v < 8; ++v) {
      const float t  = rowmax16(fmaxf(s0[v], s1[v]));
      const float mn = fmaxf(m_run[v], t);
      sc[v] = __expf(m_run[v] - mn);
      m_run[v] = mn;
      p0[v] = __expf(s0[v] - mn);
      p1[v] = __expf(s1[v] - mn);
      const float rs = rowsum16(p0[v] + p1[v]);   // UNmasked denominator
      l_run[v] = l_run[v] * sc[v] + rs;
    }
#pragma unroll
    for (int t = 0; t < 8; ++t)
#pragma unroll
      for (int v = 0; v < 8; ++v) acc[t][v] *= sc[v];

    // ---- masked probs -> LDS (C-layout scatter), re-read in A-layout ----
    __bf16* pw = &p_lds[wv * QW * PS];
#pragma unroll
    for (int v = 0; v < 8; ++v) {
      const int r = 8 * hf + v;
      pw[r * PS + ln]      = bf(p0[v] * mk0[v]);
      pw[r * PS + ln + 16] = bf(p1[v] * mk1[v]);
    }
    asm volatile("s_wait_dscnt 0" ::: "memory");  // wave-local LDS ordering

    const __bf16* pr = &p_lds[(wv * QW + ln) * PS];
    v16bf a_p = ld_pair16(pr + 16 * hf, pr + 16 * hf + 8);

    // ---- GEMM2: O += P @ V (K = 32, one WMMA per dv-tile) ----
#pragma unroll
    for (int t = 0; t < 8; ++t)
      acc[t] = __builtin_amdgcn_wmma_f32_16x16x32_bf16(false, a_p, false, bv[t], (short)0, acc[t], false, false);
  }

  // ---- epilogue: normalize by unmasked denominator, store fp32 ----
#pragma unroll
  for (int v = 0; v < 8; ++v) {
    const float inv = 1.0f / l_run[v];
    float* orow = O + ((size_t)b * S_ + (size_t)(q0 + 8 * hf + v)) * D_;
#pragma unroll
    for (int t = 0; t < 8; ++t) orow[16 * t + ln] = acc[t][v] * inv;
  }
}

extern "C" void kernel_launch(void* const* d_in, const int* in_sizes, int n_in,
                              void* d_out, int out_size, void* d_ws, size_t ws_size,
                              hipStream_t stream) {
  (void)in_sizes; (void)n_in; (void)out_size; (void)d_ws; (void)ws_size;
  const float* x1   = (const float*)d_in[0];   // Q  [B,Sq,D]
  const float* x2   = (const float*)d_in[1];   // K  [B,Skv,D]
  const float* x3   = (const float*)d_in[2];   // V  [B,Skv,Dv]
  const float* mask = (const float*)d_in[3];   // dropout mask [B,Sq,Skv]
  float* out = (float*)d_out;                  // [B,Sq,Dv]
  dim3 grid(S_ / QB, B_);
  fa_fwd<<<grid, NW * 32, 0, stream>>>(x1, x2, x3, mask, out);
}
